// VarianceAdaptorGaus_48369921687649
// MI455X (gfx1250) — matrix operand order repaired and checked by reading, not compiled, not run on hardware
//
#include <hip/hip_runtime.h>
#include <math.h>

// ---------------------------------------------------------------------------
// VarianceAdaptorGaus forward for MI455X (gfx1250), wave32 + WMMA bf16.
// B=16, T_IN=512, H=LSTM_H=FILT=256, N_BINS=256, K=3, t_out=3072.
// All GEMM B-operands are pre-swizzled into the WMMA bf16 fragment layout so
// each lane's B fragment is one contiguous 32-byte load; A fragments are
// float4-vectorized; no divergent masking in any WMMA K-loop.
// ---------------------------------------------------------------------------

typedef __attribute__((ext_vector_type(16))) __bf16 v16bf;
typedef __attribute__((ext_vector_type(8)))  float  v8f;

#define NB    16
#define TIN   512
#define TOUTX 3072
#define HDIM  256

static __device__ __forceinline__ float sigf(float x) { return 1.0f / (1.0f + __expf(-x)); }
static __device__ __forceinline__ float softplusf(float x) {
  if (x > 20.0f) return x;
  if (x < -20.0f) return __expf(x);
  return log1pf(__expf(x));
}
static __device__ __forceinline__ v8f wmma_bf16(v16bf a, v16bf b, v8f c) {
  return __builtin_amdgcn_wmma_f32_16x16x32_bf16(false, a, false, b, (short)0, c, false, false);
}
static __device__ __forceinline__ v16bf cvt16(float4 a0, float4 a1, float4 b0, float4 b1) {
  v16bf r;
  r[0]  = (__bf16)a0.x; r[1]  = (__bf16)a0.y; r[2]  = (__bf16)a0.z; r[3]  = (__bf16)a0.w;
  r[4]  = (__bf16)a1.x; r[5]  = (__bf16)a1.y; r[6]  = (__bf16)a1.z; r[7]  = (__bf16)a1.w;
  r[8]  = (__bf16)b0.x; r[9]  = (__bf16)b0.y; r[10] = (__bf16)b0.z; r[11] = (__bf16)b0.w;
  r[12] = (__bf16)b1.x; r[13] = (__bf16)b1.y; r[14] = (__bf16)b1.z; r[15] = (__bf16)b1.w;
  return r;
}
// A fragment: 8 consecutive floats at ka and 8 at ka+16 (32B-aligned runs).
static __device__ __forceinline__ v16bf load_a16(const float* p, int ka) {
  const float4* q = (const float4*)(p + ka);
  const float4* r = (const float4*)(p + ka + 16);
  return cvt16(q[0], q[1], r[0], r[1]);
}

// ------------------------- swizzle / reshape kernels -----------------------
// B-swizzle layout: Bs[((k>>5)*N + n)*32 + (k&31 with bits 3,4 swapped)] = B[k][n] as bf16.

// weights given as W[n][k] (row-major N x ldk); zero-pad k in [Kin, Kpad)
__global__ void k_swz_w(const float* __restrict__ w, __bf16* __restrict__ out,
                        int N, int Kin, int Kpad, int ldk) {
  int idx = blockIdx.x * blockDim.x + threadIdx.x;
  if (idx < N * Kpad) {
    int n = idx / Kpad, k = idx % Kpad;
    float v = (k < Kin) ? w[(size_t)n * ldk + k] : 0.0f;
    out[((((size_t)(k >> 5)) * N + n) << 5) + ((k & 7) | ((k & 8) << 1) | ((k & 16) >> 1))] = (__bf16)v;
  }
}

// conv weight (OC=256, IC=256, K=3) -> swizzled B with kk = k*256+ic, N=256
__global__ void k_swz_conv(const float* __restrict__ w, __bf16* __restrict__ out) {
  int idx = blockIdx.x * blockDim.x + threadIdx.x;
  if (idx < 256 * 768) {
    int oc = idx / 768, kk = idx % 768;
    int kidx = kk >> 8, ic = kk & 255;
    float v = w[((size_t)(oc << 8) + ic) * 3 + kidx];
    out[((((size_t)(kk >> 5)) << 8) + oc) * 32 + ((kk & 7) | ((kk & 8) << 1) | ((kk & 16) >> 1))] = (__bf16)v;
  }
}

// embeddings (B,TIN,H) -> per-batch swizzled B (K=TIN rows, N=HDIM cols)
__global__ void k_swz_emb(const float* __restrict__ emb, __bf16* __restrict__ out) {
  int idx = blockIdx.x * blockDim.x + threadIdx.x;
  if (idx < NB * TIN * HDIM) {
    int d = idx & 255, i = (idx >> 8) & 511, b = idx >> 17;
    out[((size_t)b << 17) + (((((size_t)(i >> 5)) << 8) + d) << 5) + ((i & 7) | ((i & 8) << 1) | ((i & 16) >> 1))] =
        (__bf16)emb[idx];
  }
}

// ----------------------- generic WMMA GEMM (maskless) ----------------------
// C[M,N] = A[M,K] @ Bs (+bias).  M%16==0, N%128==0, K%32==0. 8 waves/block,
// one 16x16 tile per wave.
__global__ void k_gemm(const float* __restrict__ A, const __bf16* __restrict__ Bs,
                       float* __restrict__ C, int N, int K, int lda, int ldc,
                       const float* __restrict__ bias) {
  const int lane = threadIdx.x & 31;
  const int wave = threadIdx.x >> 5;
  const int m0 = blockIdx.y << 4;
  const int n0 = (blockIdx.x << 7) + (wave << 4);
  const int col = lane & 15, hi = lane >> 4;
  const float* Ap = A + (size_t)(m0 + col) * lda;
  const int n = n0 + col;
  const __bf16* Bp = Bs + ((size_t)n << 5) + (hi << 4);
  const size_t bstep = (size_t)N << 5;
  v8f acc = {};
  for (int k0 = 0; k0 < K; k0 += 32) {
    __builtin_prefetch(Bp + bstep, 0, 1);
    v16bf a = load_a16(Ap, k0 + (hi << 3));
    v16bf b = *(const v16bf*)Bp;
    Bp += bstep;
    acc = wmma_bf16(a, b, acc);
  }
  float bv = bias ? bias[n] : 0.0f;
#pragma unroll
  for (int v = 0; v < 8; ++v)
    C[(size_t)(m0 + (hi << 3) + v) * ldc + n] = acc[v] + bv;
}

// --------------------------- LSTM recurrence -------------------------------
// One workgroup per direction. z[16,1024] = h@WhT + xw[t] via WMMA; h,c,z in
// 96KB dynamic LDS.  WhT pre-swizzled bf16 (N=1024, K=256).
__global__ void k_lstm(const float* __restrict__ xw_f, const float* __restrict__ xw_b,
                       const __bf16* __restrict__ whs_f, const __bf16* __restrict__ whs_b,
                       float* __restrict__ ys_f, float* __restrict__ ys_b, int T) {
  extern __shared__ float smem[];
  float* hS = smem;                 // 16*256
  float* cS = smem + 16 * 256;      // 16*256
  float* zS = smem + 2 * 16 * 256;  // 16*1024
  const int dir = blockIdx.x;
  const float*  xw  = dir ? xw_b  : xw_f;
  const __bf16* whs = dir ? whs_b : whs_f;
  float*        ys  = dir ? ys_b  : ys_f;
  const int lane = threadIdx.x & 31;
  const int wave = threadIdx.x >> 5;
  const int col = lane & 15, hi = lane >> 4;
  for (int i = threadIdx.x; i < 16 * 256; i += blockDim.x) { hS[i] = 0.0f; cS[i] = 0.0f; }
  __syncthreads();
  const float* hp = hS + (col << 8);
  for (int s = 0; s < T; ++s) {
    const int t = dir ? (T - 1 - s) : s;
#pragma unroll 1
    for (int nt = 0; nt < 8; ++nt) {
      const int n = (wave << 7) + (nt << 4) + col;
      v8f acc;
#pragma unroll
      for (int v = 0; v < 8; ++v) {
        int bb = (hi << 3) + v;
        acc[v] = xw[(((size_t)bb * T + t) << 10) + n];   // bias pre-folded
      }
      const __bf16* Bp = whs + ((size_t)n << 5) + (hi << 4);
#pragma unroll
      for (int k0 = 0; k0 < 256; k0 += 32) {
        v16bf a = load_a16(hp, k0 + (hi << 3));
        v16bf b = *(const v16bf*)Bp;
        Bp += (size_t)1024 << 5;
        acc = wmma_bf16(a, b, acc);
      }
#pragma unroll
      for (int v = 0; v < 8; ++v) zS[(((hi << 3) + v) << 10) + n] = acc[v];
    }
    __syncthreads();
    const int hid = threadIdx.x;            // 0..255
#pragma unroll 1
    for (int bb = 0; bb < 16; ++bb) {
      const float* zr = zS + (bb << 10);
      float iz = zr[hid], fz = zr[256 + hid], gz = zr[512 + hid], oz = zr[768 + hid];
      float cp = cS[(bb << 8) + hid];
      float cn = sigf(fz) * cp + sigf(iz) * tanhf(gz);
      float hn = sigf(oz) * tanhf(cn);
      cS[(bb << 8) + hid] = cn;
      hS[(bb << 8) + hid] = hn;
      ys[(((size_t)bb * T + t) << 8) + hid] = hn;
    }
    __syncthreads();
  }
}

// durations[b,t] = concat(ysf,ysb) . proj_w
__global__ void k_proj_dur(const float* __restrict__ ysf, const float* __restrict__ ysb,
                           const float* __restrict__ pw, float* __restrict__ out, int n) {
  int idx = blockIdx.x * blockDim.x + threadIdx.x;
  if (idx < n) {
    const float* a = ysf + (size_t)idx * 256;
    const float* b = ysb + (size_t)idx * 256;
    float s = 0.0f;
    for (int k = 0; k < 256; ++k) s += a[k] * pw[k] + b[k] * pw[256 + k];
    out[idx] = s;
  }
}

__global__ void k_ranges(const float* __restrict__ ysf, const float* __restrict__ ysb,
                         const float* __restrict__ pw, const float* __restrict__ pb,
                         float* __restrict__ out, int n) {
  int idx = blockIdx.x * blockDim.x + threadIdx.x;
  if (idx < n) {
    const float* a = ysf + (size_t)idx * 256;
    const float* b = ysb + (size_t)idx * 256;
    float s = pb[0];
    for (int k = 0; k < 256; ++k) s += a[k] * pw[k] + b[k] * pw[256 + k];
    out[idx] = softplusf(s);
  }
}

// X2 row-major 8192 x 288: [0:256)=emb, 256=durations, [257:288)=0
__global__ void k_build_x2(const float* __restrict__ emb, const float* __restrict__ dur,
                           float* __restrict__ x2) {
  int idx = blockIdx.x * blockDim.x + threadIdx.x;
  if (idx < NB * TIN * 288) {
    int r = idx / 288, c = idx % 288;
    x2[idx] = (c < 256) ? emb[(size_t)r * 256 + c] : (c == 256 ? dur[r] : 0.0f);
  }
}

// mu = cumsum(y_dur) - 0.5*y_dur ; sigma = max(ranges, eps)
__global__ void k_musigma(const float* __restrict__ ydur, const float* __restrict__ rng,
                          float* __restrict__ mu, float* __restrict__ sg) {
  int b = threadIdx.x;
  if (b < NB) {
    float cs = 0.0f;
    for (int i = 0; i < TIN; ++i) {
      float d = ydur[b * TIN + i];
      cs += d;
      mu[b * TIN + i] = cs - 0.5f * d;
      sg[b * TIN + i] = fmaxf(rng[b * TIN + i], 1e-6f);
    }
  }
}

// per (b,o): softmax max & sum over i  (constant -0.5*log(2pi) cancels)
__global__ void k_smax(const float* __restrict__ mu, const float* __restrict__ sg,
                       float* __restrict__ mx, float* __restrict__ sm) {
  int idx = blockIdx.x * blockDim.x + threadIdx.x;
  if (idx < NB * TOUTX) {
    int b = idx / TOUTX;
    float of = (float)(idx % TOUTX);
    const float* mub = mu + b * TIN;
    const float* sgb = sg + b * TIN;
    float m = -1e30f;
    for (int i = 0; i < TIN; ++i) {
      float s = sgb[i];
      float u = (of - mub[i]) / s;
      m = fmaxf(m, -0.5f * u * u - __logf(s));
    }
    float acc = 0.0f;
    for (int i = 0; i < TIN; ++i) {
      float s = sgb[i];
      float u = (of - mub[i]) / s;
      acc += __expf(-0.5f * u * u - __logf(s) - m);
    }
    mx[idx] = m;
    sm[idx] = acc;
  }
}

// x[b,o,d] = sum_i softmax_i * emb[b,i,d] + pos_enc(o,d).  Softmax weights
// generated in-register as the WMMA A-fragment; B = pre-swizzled bf16 emb.
__global__ void k_einsum(const float* __restrict__ mu, const float* __restrict__ sg,
                         const float* __restrict__ mx, const float* __restrict__ sm,
                         const __bf16* __restrict__ ebs, float* __restrict__ x) {
  const int lane = threadIdx.x & 31;
  const int wave = threadIdx.x >> 5;
  const int col = lane & 15, hi = lane >> 4;
  const int b  = blockIdx.z;
  const int o0 = ((blockIdx.y << 3) + wave) << 4;
  const int n0 = blockIdx.x << 6;
  const float* mub = mu + b * TIN;
  const float* sgb = sg + b * TIN;
  const __bf16* eb = ebs + ((size_t)b << 17);
  const int orow = o0 + col;
  const float of  = (float)orow;
  const float m   = mx[b * TOUTX + orow];
  const float inv = 1.0f / sm[b * TOUTX + orow];
  const __bf16* Bp0 = eb + ((size_t)(n0 + col) << 5) + (hi << 4);
  v8f acc[4] = {{}, {}, {}, {}};
  for (int k0 = 0; k0 < TIN; k0 += 32) {
    v16bf a;
    const int ka = k0 + (hi << 3);
#pragma unroll
    for (int e = 0; e < 8; ++e) {
      { int i = ka + e;      float s = sgb[i]; float u = (of - mub[i]) / s;
        a[e]     = (__bf16)(__expf(-0.5f * u * u - __logf(s) - m) * inv); }
      { int i = ka + 16 + e; float s = sgb[i]; float u = (of - mub[i]) / s;
        a[e + 8] = (__bf16)(__expf(-0.5f * u * u - __logf(s) - m) * inv); }
    }
    const __bf16* Bp = Bp0 + ((size_t)(k0 >> 5) << 13);   // + (k0/32)*256*32
#pragma unroll
    for (int tt = 0; tt < 4; ++tt) {
      v16bf bf = *(const v16bf*)(Bp + ((size_t)tt << 9)); // +16 cols * 32
      acc[tt] = wmma_bf16(a, bf, acc[tt]);
    }
  }
  float* xb = x + (size_t)b * TOUTX * HDIM;
#pragma unroll
  for (int v = 0; v < 8; ++v) {
    const int o = o0 + (hi << 3) + v;
    const float ofo = (float)o;
#pragma unroll
    for (int tt = 0; tt < 4; ++tt) {
      const int d = n0 + (tt << 4) + col;
      float freq = __expf(-9.210340371976184f * (float)(d & ~1) * (1.0f / 256.0f));
      float pe = (d & 1) ? __cosf(ofo * freq) : __sinf(ofo * freq);
      xb[(size_t)o * HDIM + d] = acc[tt][v] + pe;
    }
  }
}

// ------------------------------ conv as GEMM -------------------------------
// y[b,t,oc] = relu(bias + sum_{k,ic} w[oc,ic,k]*x[b,t+k-1,ic]); K=768.
// Interior tiles: pure float4 loads (8-elem A runs never cross a kpos edge).
static __device__ __forceinline__ float4 ld4z(const float* base, int st, int ic) {
  if (st >= 0 && st < TOUTX) return *(const float4*)(base + ((size_t)st << 8) + ic);
  return make_float4(0.0f, 0.0f, 0.0f, 0.0f);
}
__global__ void k_conv(const float* __restrict__ xin, const __bf16* __restrict__ Bs,
                       const float* __restrict__ bias, float* __restrict__ yout) {
  const int lane = threadIdx.x & 31;
  const int wave = threadIdx.x >> 5;
  const int col = lane & 15, hi = lane >> 4;
  const int b  = blockIdx.y;
  const int t0 = blockIdx.x << 4;
  const int n0 = wave << 5;
  const float* xb = xin + (size_t)b * TOUTX * HDIM;
  const int arow = t0 + col;
  const bool interior = (t0 >= 1) && (t0 <= TOUTX - 17);   // block-uniform
  const __bf16* Bp0 = Bs + ((size_t)(n0 + col) << 5) + (hi << 4);
  const __bf16* Bp1 = Bp0 + (16 << 5);
  v8f acc0 = {}, acc1 = {};
#pragma unroll 1
  for (int k0 = 0; k0 < 768; k0 += 32) {
    const int ka = k0 + (hi << 3);
    const int kk0 = ka, kk1 = ka + 16;
    const int st0 = arow + (kk0 >> 8) - 1, ic0 = kk0 & 255;
    const int st1 = arow + (kk1 >> 8) - 1, ic1 = kk1 & 255;
    v16bf a;
    if (interior) {
      const float4* q = (const float4*)(xb + ((size_t)st0 << 8) + ic0);
      const float4* r = (const float4*)(xb + ((size_t)st1 << 8) + ic1);
      a = cvt16(q[0], q[1], r[0], r[1]);
    } else {
      a = cvt16(ld4z(xb, st0, ic0), ld4z(xb, st0, ic0 + 4),
                ld4z(xb, st1, ic1), ld4z(xb, st1, ic1 + 4));
    }
    const size_t koff = (size_t)(k0 >> 5) << 13;           // (k0/32)*256*32
    v16bf b0 = *(const v16bf*)(Bp0 + koff);
    v16bf b1 = *(const v16bf*)(Bp1 + koff);
    acc0 = wmma_bf16(a, b0, acc0);
    acc1 = wmma_bf16(a, b1, acc1);
  }
  float* yb = yout + (size_t)b * TOUTX * HDIM;
  const float bv0 = bias[n0 + col], bv1 = bias[n0 + 16 + col];
#pragma unroll
  for (int v = 0; v < 8; ++v) {
    const int t = t0 + (hi << 3) + v;
    yb[((size_t)t << 8) + n0 + col]      = fmaxf(acc0[v] + bv0, 0.0f);
    yb[((size_t)t << 8) + n0 + 16 + col] = fmaxf(acc1[v] + bv1, 0.0f);
  }
}

// LayerNorm over 256 channels, one wave32 per row, shuffle reduction.
__global__ void k_ln(float* __restrict__ h, const float* __restrict__ g,
                     const float* __restrict__ be, int nrows) {
  int row = blockIdx.x * 8 + (threadIdx.x >> 5);
  int lane = threadIdx.x & 31;
  if (row >= nrows) return;
  float* p = h + (size_t)row * 256;
  float vals[8];
  float s = 0.0f, s2 = 0.0f;
#pragma unroll
  for (int j = 0; j < 8; ++j) { float v = p[lane + j * 32]; vals[j] = v; s += v; s2 += v * v; }
#pragma unroll
  for (int off = 16; off > 0; off >>= 1) {
    s  += __shfl_xor(s, off, 32);
    s2 += __shfl_xor(s2, off, 32);
  }
  float mean = s * (1.0f / 256.0f);
  float var  = s2 * (1.0f / 256.0f) - mean * mean;
  float rstd = rsqrtf(var + 1e-5f);
#pragma unroll
  for (int j = 0; j < 8; ++j) {
    int d = lane + j * 32;
    p[d] = (vals[j] - mean) * rstd * g[d] + be[d];
  }
}

__global__ void k_vpout(const float* __restrict__ h, const float* __restrict__ lw,
                        const float* __restrict__ lb, const unsigned char* __restrict__ mask,
                        float* __restrict__ out, int n) {
  int idx = blockIdx.x * blockDim.x + threadIdx.x;
  if (idx < n) {
    const float* p = h + (size_t)idx * 256;
    float s = lb[0];
    for (int k = 0; k < 256; ++k) s += p[k] * lw[k];
    out[idx] = mask[idx] ? 0.0f : s;
  }
}

// x[row,:] += emb[searchsorted(bins, target[row]), :]
__global__ void k_addemb(float* __restrict__ x, const float* __restrict__ emb,
                         const float* __restrict__ bins, const float* __restrict__ tgt) {
  int row = blockIdx.x;
  int d = threadIdx.x;
  float v = tgt[row];
  int lo = 0, hiB = 255;
  while (lo < hiB) { int mid = (lo + hiB) >> 1; if (bins[mid] < v) lo = mid + 1; else hiB = mid; }
  x[(size_t)row * 256 + d] += emb[(size_t)lo * 256 + d];
}

// ------------------------------ host driver --------------------------------

extern "C" void kernel_launch(void* const* d_in, const int* in_sizes, int n_in,
                              void* d_out, int out_size, void* d_ws, size_t ws_size,
                              hipStream_t stream) {
  (void)in_sizes; (void)n_in; (void)out_size; (void)ws_size;
  const float* emb  = (const float*)d_in[0];
  const float* ydur = (const float*)d_in[1];
  const float* ptgt = (const float*)d_in[2];
  const float* etgt = (const float*)d_in[3];
  const unsigned char* mask = (const unsigned char*)d_in[4];
  // d_in[5] = input_lengths (unused)
  // params: jax pytree alphabetical flatten
  const float* dur_b_Wh = (const float*)d_in[6];
  const float* dur_b_Wi = (const float*)d_in[7];
  const float* dur_b_bi = (const float*)d_in[8];
  const float* dur_f_Wh = (const float*)d_in[9];
  const float* dur_f_Wi = (const float*)d_in[10];
  const float* dur_f_bi = (const float*)d_in[11];
  const float* dur_pw   = (const float*)d_in[12];
  const float* e_bins   = (const float*)d_in[13];
  const float* e_embT   = (const float*)d_in[14];
  const float* evp_b1   = (const float*)d_in[15];
  const float* evp_b2   = (const float*)d_in[16];
  const float* evp_be1  = (const float*)d_in[17];
  const float* evp_be2  = (const float*)d_in[18];
  const float* evp_g1   = (const float*)d_in[19];
  const float* evp_g2   = (const float*)d_in[20];
  const float* evp_lb   = (const float*)d_in[21];
  const float* evp_lw   = (const float*)d_in[22];
  const float* evp_w1   = (const float*)d_in[23];
  const float* evp_w2   = (const float*)d_in[24];
  const float* p_bins   = (const float*)d_in[25];
  const float* p_embT   = (const float*)d_in[26];
  const float* pvp_b1   = (const float*)d_in[27];
  const float* pvp_b2   = (const float*)d_in[28];
  const float* pvp_be1  = (const float*)d_in[29];
  const float* pvp_be2  = (const float*)d_in[30];
  const float* pvp_g1   = (const float*)d_in[31];
  const float* pvp_g2   = (const float*)d_in[32];
  const float* pvp_lb   = (const float*)d_in[33];
  const float* pvp_lw   = (const float*)d_in[34];
  const float* pvp_w1   = (const float*)d_in[35];
  const float* pvp_w2   = (const float*)d_in[36];
  const float* rng_b_Wh = (const float*)d_in[37];
  const float* rng_b_Wi = (const float*)d_in[38];
  const float* rng_b_bi = (const float*)d_in[39];
  const float* rng_f_Wh = (const float*)d_in[40];
  const float* rng_f_Wi = (const float*)d_in[41];
  const float* rng_f_bi = (const float*)d_in[42];
  const float* rng_pb   = (const float*)d_in[43];
  const float* rng_pw   = (const float*)d_in[44];

  // output layout: x | durations | pitch_pred | energy_pred
  float* out_x      = (float*)d_out;
  float* out_dur    = out_x + (size_t)NB * TOUTX * HDIM;   // 12,582,912
  float* out_pitch  = out_dur + NB * TIN;
  float* out_energy = out_pitch + NB * TOUTX;

  // workspace layout (floats). Region A reused across phases.
  float* ws = (float*)d_ws;
  float* XWf = ws;                         // 8,388,608  (also XW2f)
  float* XWb = ws + 8388608;               // 8,388,608  (also XW2b)
  float* ysf = ws + 16777216;              // 2,097,152  (also ys2f)
  float* ysb = ws + 18874368;              // 2,097,152  (also ys2b)
  float* X2  = ws + 20971520;              // 8192*288 = 2,359,296
  float* h1  = ws;                         // 12,582,912 (VP phase reuse)
  float* h2  = ws + 12582912;              // 12,582,912
  size_t off = 25165824;
  float* ranges = ws + off; off += 8192;
  float* muB    = ws + off; off += 8192;
  float* sgB    = ws + off; off += 8192;
  float* smM    = ws + off; off += 49152;
  float* smS    = ws + off; off += 49152;
  off = (off + 15) & ~(size_t)15;          // 64B-align bf16 region
  __bf16* bws = (__bf16*)(ws + off);
  size_t boff = 0;
  auto balloc = [&](size_t n) { __bf16* p = bws + boff; boff += (n + 31) & ~(size_t)31; return p; };
  __bf16* WiS_df = balloc(256 * 1024);
  __bf16* WiS_db = balloc(256 * 1024);
  __bf16* WiS_rf = balloc(288 * 1024);
  __bf16* WiS_rb = balloc(288 * 1024);
  __bf16* WhS_df = balloc(256 * 1024);
  __bf16* WhS_db = balloc(256 * 1024);
  __bf16* WhS_rf = balloc(256 * 1024);
  __bf16* WhS_rb = balloc(256 * 1024);
  __bf16* CW_p1  = balloc(768 * 256);
  __bf16* CW_p2  = balloc(768 * 256);
  __bf16* CW_e1  = balloc(768 * 256);
  __bf16* CW_e2  = balloc(768 * 256);
  __bf16* ebs    = balloc((size_t)NB * TIN * HDIM);

  // --- weight / operand swizzles (once per launch, cheap) ---
  k_swz_w<<<1024, 256, 0, stream>>>(dur_f_Wi, WiS_df, 1024, 256, 256, 256);
  k_swz_w<<<1024, 256, 0, stream>>>(dur_b_Wi, WiS_db, 1024, 256, 256, 256);
  k_swz_w<<<1152, 256, 0, stream>>>(rng_f_Wi, WiS_rf, 1024, 257, 288, 257);
  k_swz_w<<<1152, 256, 0, stream>>>(rng_b_Wi, WiS_rb, 1024, 257, 288, 257);
  k_swz_w<<<1024, 256, 0, stream>>>(dur_f_Wh, WhS_df, 1024, 256, 256, 256);
  k_swz_w<<<1024, 256, 0, stream>>>(dur_b_Wh, WhS_db, 1024, 256, 256, 256);
  k_swz_w<<<1024, 256, 0, stream>>>(rng_f_Wh, WhS_rf, 1024, 256, 256, 256);
  k_swz_w<<<1024, 256, 0, stream>>>(rng_b_Wh, WhS_rb, 1024, 256, 256, 256);
  k_swz_conv<<<768, 256, 0, stream>>>(pvp_w1, CW_p1);
  k_swz_conv<<<768, 256, 0, stream>>>(pvp_w2, CW_p2);
  k_swz_conv<<<768, 256, 0, stream>>>(evp_w1, CW_e1);
  k_swz_conv<<<768, 256, 0, stream>>>(evp_w2, CW_e2);
  k_swz_emb<<<8192, 256, 0, stream>>>(emb, ebs);

  // --- BiLSTM #1 (durations) ---
  dim3 gxw(8, 512);  // N=1024, M=8192
  k_gemm<<<gxw, 256, 0, stream>>>(emb, WiS_df, XWf, 1024, 256, 256, 1024, dur_f_bi);
  k_gemm<<<gxw, 256, 0, stream>>>(emb, WiS_db, XWb, 1024, 256, 256, 1024, dur_b_bi);
  k_lstm<<<2, 256, 98304, stream>>>(XWf, XWb, WhS_df, WhS_db, ysf, ysb, TIN);
  k_proj_dur<<<32, 256, 0, stream>>>(ysf, ysb, dur_pw, out_dur, NB * TIN);

  // --- BiLSTM #2 (ranges) on concat(emb, durations), K padded to 288 ---
  k_build_x2<<<(NB * TIN * 288 + 255) / 256, 256, 0, stream>>>(emb, out_dur, X2);
  k_gemm<<<gxw, 256, 0, stream>>>(X2, WiS_rf, XWf, 1024, 288, 288, 1024, rng_f_bi);
  k_gemm<<<gxw, 256, 0, stream>>>(X2, WiS_rb, XWb, 1024, 288, 288, 1024, rng_b_bi);
  k_lstm<<<2, 256, 98304, stream>>>(XWf, XWb, WhS_rf, WhS_rb, ysf, ysb, TIN);
  k_ranges<<<32, 256, 0, stream>>>(ysf, ysb, rng_pw, rng_pb, ranges, NB * TIN);

  // --- Gaussian upsampling (fused softmax + einsum + pos-enc) ---
  k_musigma<<<1, 16, 0, stream>>>(ydur, ranges, muB, sgB);
  k_smax<<<192, 256, 0, stream>>>(muB, sgB, smM, smS);
  k_einsum<<<dim3(4, 24, 16), 256, 0, stream>>>(muB, sgB, smM, smS, ebs, out_x);

  // --- pitch variance predictor + pitch embedding add ---
  dim3 gc(TOUTX / 16, NB);
  k_conv<<<gc, 256, 0, stream>>>(out_x, CW_p1, pvp_b1, h1);
  k_ln<<<6144, 256, 0, stream>>>(h1, pvp_g1, pvp_be1, NB * TOUTX);
  k_conv<<<gc, 256, 0, stream>>>(h1, CW_p2, pvp_b2, h2);
  k_ln<<<6144, 256, 0, stream>>>(h2, pvp_g2, pvp_be2, NB * TOUTX);
  k_vpout<<<192, 256, 0, stream>>>(h2, pvp_lw, pvp_lb, mask, out_pitch, NB * TOUTX);
  k_addemb<<<NB * TOUTX, 256, 0, stream>>>(out_x, p_embT, p_bins, ptgt);

  // --- energy variance predictor + energy embedding add ---
  k_conv<<<gc, 256, 0, stream>>>(out_x, CW_e1, evp_b1, h1);
  k_ln<<<6144, 256, 0, stream>>>(h1, evp_g1, evp_be1, NB * TOUTX);
  k_conv<<<gc, 256, 0, stream>>>(h1, CW_e2, evp_b2, h2);
  k_ln<<<6144, 256, 0, stream>>>(h2, evp_g2, evp_be2, NB * TOUTX);
  k_vpout<<<192, 256, 0, stream>>>(h2, evp_lw, evp_lb, mask, out_energy, NB * TOUTX);
  k_addemb<<<NB * TOUTX, 256, 0, stream>>>(out_x, e_embT, e_bins, etgt);
}
